// Attention_27530740367526
// MI455X (gfx1250) — matrix operand order repaired and checked by reading, not compiled, not run on hardware
//
#include <hip/hip_runtime.h>

// ---------------- problem constants (from reference) ----------------
#define BATCH   2
#define SEQ     2048
#define D_MODEL 2048
#define HEADS   16
#define HD      128        // head dim

typedef __attribute__((ext_vector_type(16))) __bf16 v16bf;
typedef __attribute__((ext_vector_type(8)))  float  v8f;

union FragA { v16bf v; unsigned int u[8]; };

__device__ __forceinline__ unsigned short f2bf(float f) {
  unsigned int u = __float_as_uint(f);
  unsigned int r = u + 0x7FFFu + ((u >> 16) & 1u);   // round-to-nearest-even
  return (unsigned short)(r >> 16);
}
__device__ __forceinline__ float bf2f(unsigned short h) {
  return __uint_as_float(((unsigned int)h) << 16);
}
__device__ __forceinline__ v8f vzero8() {
  v8f z;
  #pragma unroll
  for (int i = 0; i < 8; ++i) z[i] = 0.0f;
  return z;
}

// gfx1250 async global->LDS copy: 16B per lane, tracked by ASYNCcnt.
// lds_off = wave-relative LDS byte offset (low 32 bits of generic LDS pointer).
__device__ __forceinline__ void async_copy_b128(unsigned lds_off,
                                                const unsigned short* gaddr) {
  asm volatile("global_load_async_to_lds_b128 %0, %1, off"
               :: "v"(lds_off), "v"(gaddr) : "memory");
}
__device__ __forceinline__ void wait_asynccnt0() {
  asm volatile("s_wait_asynccnt 0" ::: "memory");
}

// ---------------- fp32 -> bf16 conversion ----------------
__global__ void cvt_bf16_kernel(const float* __restrict__ in,
                                unsigned short* __restrict__ out, long n) {
  long i = (long)blockIdx.x * blockDim.x + threadIdx.x;
  if (i < n) out[i] = f2bf(in[i]);
}

// ---------------- bf16 WMMA GEMM:  C[M,N] = A[M,K] * B[N,K]^T ----------------
// A row-major [M][K], B row-major [N][K] (== torch Linear weight), both bf16.
// Block: 256 threads (8 waves, 2x4), tile 128(M) x 128(N), BK=32.
template<int OUTF32>
__global__ __launch_bounds__(256)
void gemm_bf16_kernel(const unsigned short* __restrict__ A,
                      const unsigned short* __restrict__ Bw,
                      void* __restrict__ Cout,
                      int M, int N, int K) {
  __shared__ unsigned short lds_a[128 * 40];   // pad stride 40 shorts (80B, 16B-aligned)
  __shared__ unsigned short lds_b[128 * 40];

  const int tid   = threadIdx.x;
  const int lane  = tid & 31;
  const int wave  = tid >> 5;
  const int waveM = wave >> 2;        // 0..1  -> 64 rows each
  const int waveN = wave & 3;         // 0..3  -> 32 cols each
  const int half  = lane >> 4;        // 0/1
  const int ln16  = lane & 15;

  const long rowBase = (long)blockIdx.y * 128;
  const long colBase = (long)blockIdx.x * 128;

  v8f acc[4][2];
  #pragma unroll
  for (int r = 0; r < 4; ++r)
    #pragma unroll
    for (int c = 0; c < 2; ++c) acc[r][c] = vzero8();

  for (int k0 = 0; k0 < K; k0 += 32) {
    // async stage of A(128x32) and B(128x32) tiles straight into LDS:
    // 512 x 16B segments per tile, 2 segments per thread per tile.
    #pragma unroll
    for (int it = 0; it < 2; ++it) {
      int idx = tid + it * 256;           // 0..511
      int row = idx >> 2;                 // 0..127
      int seg = idx & 3;                  // 8 bf16 per segment
      unsigned la = (unsigned)(size_t)&lds_a[row * 40 + seg * 8];
      unsigned lb = (unsigned)(size_t)&lds_b[row * 40 + seg * 8];
      async_copy_b128(la, A  + (rowBase + row) * (long)K + k0 + seg * 8);
      async_copy_b128(lb, Bw + (colBase + row) * (long)K + k0 + seg * 8);
    }
    wait_asynccnt0();                     // our async writes are in LDS
    __syncthreads();                      // everyone's writes visible

    if (k0 + 32 < K) {                    // advisory prefetch of next K-chunk
      int prow = tid & 127;
      __builtin_prefetch(A  + (rowBase + prow) * (long)K + k0 + 32, 0, 1);
      __builtin_prefetch(Bw + (colBase + prow) * (long)K + k0 + 32, 0, 1);
    }

    // B fragments (32x16, K-dim = k-chunk): lane n = col, K pair base = 2i + 16*half
    FragA bfrag[2];
    #pragma unroll
    for (int c = 0; c < 2; ++c) {
      int n = waveN * 32 + c * 16 + ln16;
      const unsigned int* lp = (const unsigned int*)(&lds_b[n * 40]);
      #pragma unroll
      for (int i = 0; i < 8; ++i) bfrag[c].u[i] = lp[i + 8 * half];
    }
    // A fragments (16x32): lane m = row, K pair base = 2i + 16*half
    #pragma unroll
    for (int r = 0; r < 4; ++r) {
      FragA afrag;
      int m = waveM * 64 + r * 16 + ln16;
      const unsigned int* lp = (const unsigned int*)(&lds_a[m * 40]);
      #pragma unroll
      for (int i = 0; i < 8; ++i) afrag.u[i] = lp[i + 8 * half];
      #pragma unroll
      for (int c = 0; c < 2; ++c)
        acc[r][c] = __builtin_amdgcn_wmma_f32_16x16x32_bf16(
            false, afrag.v, false, bfrag[c].v, (short)0, acc[r][c], false, false);
    }
    __syncthreads();
  }

  // C layout: lane holds col n = ln16, rows m = e + 8*half (e = vgpr index)
  #pragma unroll
  for (int r = 0; r < 4; ++r)
    #pragma unroll
    for (int c = 0; c < 2; ++c) {
      long col = colBase + waveN * 32 + c * 16 + ln16;
      #pragma unroll
      for (int e = 0; e < 8; ++e) {
        long row = rowBase + waveM * 64 + r * 16 + e + 8 * half;
        float v = acc[r][c][e];
        if (OUTF32) ((float*)Cout)[row * N + col] = v;
        else        ((unsigned short*)Cout)[row * N + col] = f2bf(v);
      }
    }
}

// ---------------- RoPE: bf16 [B,S,D] -> bf16 [B,H,S,HD] ----------------
__global__ void rope_kernel(const unsigned short* __restrict__ X,
                            const float* __restrict__ Cos,
                            const float* __restrict__ Sin,
                            unsigned short* __restrict__ Outp) {
  long idx = (long)blockIdx.x * blockDim.x + threadIdx.x;
  const long total = (long)BATCH * SEQ * HEADS * (HD / 2);
  if (idx >= total) return;
  int  d = (int)(idx & 63);
  long t = idx >> 6;
  int  h = (int)(t & 15);  t >>= 4;
  int  s = (int)(t & (SEQ - 1)); t >>= 11;
  int  b = (int)t;
  long src = ((long)(b * SEQ + s)) * D_MODEL + h * HD;
  float x0 = bf2f(X[src + d]);
  float x1 = bf2f(X[src + d + 64]);
  float c0 = Cos[s * HD + d], c1 = Cos[s * HD + d + 64];
  float s0 = Sin[s * HD + d], s1 = Sin[s * HD + d + 64];
  float o0 = x0 * c0 - x1 * s0;          // x*cos + rotate_half(x)*sin, first half
  float o1 = x1 * c1 + x0 * s1;          // second half
  long dst = (((long)(b * HEADS + h)) * SEQ + s) * HD;
  Outp[dst + d]      = f2bf(o0);
  Outp[dst + d + 64] = f2bf(o1);
}

// ---------------- V transpose: bf16 [B,S,D] -> bf16 [B,H,HD,S] ----------------
__global__ void vtrans_kernel(const unsigned short* __restrict__ V,
                              unsigned short* __restrict__ Vt) {
  long idx = (long)blockIdx.x * blockDim.x + threadIdx.x;
  const long total = (long)BATCH * SEQ * D_MODEL;
  if (idx >= total) return;
  int  s = (int)(idx & (SEQ - 1));
  long t = idx >> 11;
  int  d = (int)(t & 127); t >>= 7;
  int  h = (int)(t & 15);  t >>= 4;
  int  b = (int)t;
  Vt[idx] = V[((long)(b * SEQ + s)) * D_MODEL + h * HD + d];
}

// ---------------- flash attention, one wave per 16-row query block ----------------
// Q,K: [B,H,S,HD] bf16 (roped); Vt: [B,H,HD,S] bf16; Out: [B,S,D] bf16
__global__ __launch_bounds__(128)
void attn_kernel(const unsigned short* __restrict__ Q,
                 const unsigned short* __restrict__ Kk,
                 const unsigned short* __restrict__ Vt,
                 unsigned short* __restrict__ Out) {
  __shared__ unsigned short ldsP[4][16 * 36];   // per-wave P tile, stride 36 shorts

  const int lane = threadIdx.x & 31;
  const int wave = threadIdx.x >> 5;
  const int half = lane >> 4;
  const int ln16 = lane & 15;
  const int qb   = blockIdx.x * 4 + wave;       // 16-row query block index
  const int h    = blockIdx.y;
  const int b    = blockIdx.z;
  const int qBase = qb * 16;

  const long qkBase = ((long)(b * HEADS + h)) * SEQ * HD;
  const long vBase  = ((long)(b * HEADS + h)) * HD * SEQ;
  const unsigned short* q  = Q  + qkBase;
  const unsigned short* k  = Kk + qkBase;
  const unsigned short* vt = Vt + vBase;

  // Q fragments: 16 rows x full HD=128 as four 16x32 A-fragments
  FragA qa[4];
  {
    const unsigned int* qp =
        (const unsigned int*)(q + (long)(qBase + ln16) * HD);
    #pragma unroll
    for (int t = 0; t < 4; ++t)
      #pragma unroll
      for (int i = 0; i < 8; ++i)
        qa[t].u[i] = qp[16 * t + i + 8 * half];
  }

  v8f o[8];
  #pragma unroll
  for (int c = 0; c < 8; ++c) o[c] = vzero8();
  float rm[8], rl[8];
  #pragma unroll
  for (int e = 0; e < 8; ++e) { rm[e] = -1e30f; rl[e] = 0.0f; }

  const float scale = 0.08838834764831845f;     // 1/sqrt(128)
  const int kbCount = (qBase + 16 + 31) / 32;   // causal: keys <= qBase+15

  for (int kb = 0; kb < kbCount; ++kb) {
    const int kBase = kb * 32;

    // scores S[16 x 32] = Q * K^T  (two 16x16 C tiles, 4 hd-chunks each)
    v8f s0 = vzero8(), s1 = vzero8();
    const unsigned int* kp0 = (const unsigned int*)(k + (long)(kBase + ln16) * HD);
    const unsigned int* kp1 = (const unsigned int*)(k + (long)(kBase + 16 + ln16) * HD);
    #pragma unroll
    for (int ch = 0; ch < 4; ++ch) {
      FragA kf0, kf1;
      #pragma unroll
      for (int i = 0; i < 8; ++i) {
        kf0.u[i] = kp0[16 * ch + i + 8 * half];
        kf1.u[i] = kp1[16 * ch + i + 8 * half];
      }
      s0 = __builtin_amdgcn_wmma_f32_16x16x32_bf16(
          false, qa[ch].v, false, kf0.v, (short)0, s0, false, false);
      s1 = __builtin_amdgcn_wmma_f32_16x16x32_bf16(
          false, qa[ch].v, false, kf1.v, (short)0, s1, false, false);
    }

    // online softmax over the 32 keys of this block
    const int key0 = kBase + ln16;
    const int key1 = kBase + 16 + ln16;
    #pragma unroll
    for (int e = 0; e < 8; ++e) {
      int row = qBase + e + 8 * half;
      float v0 = s0[e] * scale + (key0 <= row ? 0.0f : -1e30f);
      float v1 = s1[e] * scale + (key1 <= row ? 0.0f : -1e30f);
      float mx = fmaxf(v0, v1);
      mx = fmaxf(mx, __shfl_xor(mx, 1, 32));
      mx = fmaxf(mx, __shfl_xor(mx, 2, 32));
      mx = fmaxf(mx, __shfl_xor(mx, 4, 32));
      mx = fmaxf(mx, __shfl_xor(mx, 8, 32));
      float nm    = fmaxf(rm[e], mx);
      float alpha = __expf(rm[e] - nm);
      float p0    = __expf(v0 - nm);
      float p1    = __expf(v1 - nm);
      float rs = p0 + p1;
      rs += __shfl_xor(rs, 1, 32);
      rs += __shfl_xor(rs, 2, 32);
      rs += __shfl_xor(rs, 4, 32);
      rs += __shfl_xor(rs, 8, 32);
      rl[e] = rl[e] * alpha + rs;
      rm[e] = nm;
      #pragma unroll
      for (int c = 0; c < 8; ++c) o[c][e] *= alpha;
      // stash P (C-layout) into LDS for re-load in A-layout
      ldsP[wave][(e + 8 * half) * 36 + ln16]      = f2bf(p0);
      ldsP[wave][(e + 8 * half) * 36 + 16 + ln16] = f2bf(p1);
    }

    asm volatile("s_wait_dscnt 0" ::: "memory");

    // P as 16x32 A-fragment
    FragA pf;
    {
      const unsigned int* lp = (const unsigned int*)(&ldsP[wave][ln16 * 36]);
      #pragma unroll
      for (int i = 0; i < 8; ++i) pf.u[i] = lp[i + 8 * half];
    }

    // O += P * V : eight 16x16 output tiles over hd
    #pragma unroll
    for (int c = 0; c < 8; ++c) {
      FragA vf;
      const unsigned int* vp =
          (const unsigned int*)(vt + (long)(c * 16 + ln16) * SEQ + kBase);
      #pragma unroll
      for (int i = 0; i < 8; ++i) vf.u[i] = vp[i + 8 * half];
      o[c] = __builtin_amdgcn_wmma_f32_16x16x32_bf16(
          false, pf.v, false, vf.v, (short)0, o[c], false, false);
    }
  }

  // normalize and write [B,S,D] bf16 (D index = h*HD + hd)
  const long outRowBase = (long)b * SEQ * D_MODEL;
  #pragma unroll
  for (int e = 0; e < 8; ++e) {
    float inv = 1.0f / rl[e];
    int row = qBase + e + 8 * half;
    #pragma unroll
    for (int c = 0; c < 8; ++c) {
      int col = h * HD + c * 16 + ln16;
      Out[outRowBase + (long)row * D_MODEL + col] = f2bf(o[c][e] * inv);
    }
  }
}

// ---------------- host-side orchestration ----------------
extern "C" void kernel_launch(void* const* d_in, const int* in_sizes, int n_in,
                              void* d_out, int out_size, void* d_ws, size_t ws_size,
                              hipStream_t stream) {
  (void)in_sizes; (void)n_in; (void)out_size; (void)ws_size;
  const float* x    = (const float*)d_in[0];
  const float* cosp = (const float*)d_in[1];
  const float* sinp = (const float*)d_in[2];
  // d_in[3] = mask : causal mask computed analytically in attn_kernel
  const float* wq   = (const float*)d_in[4];
  const float* wk   = (const float*)d_in[5];
  const float* wv   = (const float*)d_in[6];
  const float* wo   = (const float*)d_in[7];

  const long MD = (long)BATCH * SEQ * D_MODEL;   // 8,388,608
  const long WD = (long)D_MODEL * D_MODEL;       // 4,194,304
  const int  M  = BATCH * SEQ;                   // 4096

  unsigned short* ws  = (unsigned short*)d_ws;
  unsigned short* xb  = ws;            // bf16 x            [B,S,D]
  unsigned short* wqb = xb  + MD;      // bf16 weights      [D,D] x4
  unsigned short* wkb = wqb + WD;
  unsigned short* wvb = wkb + WD;
  unsigned short* wob = wvb + WD;
  unsigned short* qb  = wob + WD;      // bf16 q,k,v        [B,S,D]
  unsigned short* kb  = qb  + MD;
  unsigned short* vb  = kb  + MD;
  unsigned short* qr  = vb  + MD;      // roped q,k         [B,H,S,HD]
  unsigned short* kr  = qr  + MD;
  unsigned short* vt  = kr  + MD;      // v transposed      [B,H,HD,S]
  unsigned short* attn= vt  + MD;      // attention output  [B,S,D]

  // 1) downconvert activations + weights to bf16
  cvt_bf16_kernel<<<(unsigned)((MD + 255) / 256), 256, 0, stream>>>(x,  xb,  MD);
  cvt_bf16_kernel<<<(unsigned)((WD + 255) / 256), 256, 0, stream>>>(wq, wqb, WD);
  cvt_bf16_kernel<<<(unsigned)((WD + 255) / 256), 256, 0, stream>>>(wk, wkb, WD);
  cvt_bf16_kernel<<<(unsigned)((WD + 255) / 256), 256, 0, stream>>>(wv, wvb, WD);
  cvt_bf16_kernel<<<(unsigned)((WD + 255) / 256), 256, 0, stream>>>(wo, wob, WD);

  // 2) QKV projections: [4096,2048] x [2048,2048]^T
  dim3 gg(D_MODEL / 128, M / 128);     // (16, 32)
  gemm_bf16_kernel<0><<<gg, 256, 0, stream>>>(xb, wqb, qb, M, D_MODEL, D_MODEL);
  gemm_bf16_kernel<0><<<gg, 256, 0, stream>>>(xb, wkb, kb, M, D_MODEL, D_MODEL);
  gemm_bf16_kernel<0><<<gg, 256, 0, stream>>>(xb, wvb, vb, M, D_MODEL, D_MODEL);

  // 3) RoPE + V transpose
  const long RP = (long)BATCH * SEQ * HEADS * (HD / 2);
  rope_kernel<<<(unsigned)((RP + 255) / 256), 256, 0, stream>>>(qb, cosp, sinp, qr);
  rope_kernel<<<(unsigned)((RP + 255) / 256), 256, 0, stream>>>(kb, cosp, sinp, kr);
  vtrans_kernel<<<(unsigned)((MD + 255) / 256), 256, 0, stream>>>(vb, vt);

  // 4) causal flash attention
  dim3 ga(SEQ / 16 / 4, HEADS, BATCH); // (32, 16, 2), 128 threads = 4 waves
  attn_kernel<<<ga, 128, 0, stream>>>(qr, kr, vt, attn);

  // 5) output projection -> fp32 d_out
  gemm_bf16_kernel<1><<<gg, 256, 0, stream>>>(attn, wob, d_out, M, D_MODEL, D_MODEL);
}
